// ODEBlock_26036091748409
// MI455X (gfx1250) — matrix operand order repaired and checked by reading, not compiled, not run on hardware
//
#include <hip/hip_runtime.h>

// ---- CDNA5 (gfx1250) wave32 WMMA implementation of the Neural-ODE block ----
// y0 = x@W_in + b_in ; 32x RK4 steps of f(y)=tanh(y@Wf+bf) ; out = y1@W_out + b_out
// One wave = one 16-row batch tile, fully register/LDS resident.
// WMMA chains start from inline-0 C (no D=C register copies); biases are added
// in the VALU epilogue where they fuse into the tanh dependency chain.
// f32->f16 for the LDS relayout is done as packed row-pairs (v_cvt_pk_f16_f32)
// into independent registers; odd rows use ds_store_b16_d16_hi.

typedef _Float16 v16h __attribute__((ext_vector_type(16)));
typedef _Float16 h8   __attribute__((ext_vector_type(8)));
typedef _Float16 h2   __attribute__((ext_vector_type(2)));
typedef float    v8f  __attribute__((ext_vector_type(8)));
typedef float    f2   __attribute__((ext_vector_type(2)));

#if __has_builtin(__builtin_amdgcn_tanhf)
#define TANHF(v) __builtin_amdgcn_tanhf(v)   // V_TANH_F32 hardware transcendental
#else
#define TANHF(v) tanhf(v)
#endif

#define ROWH 72   // scratch row stride in halves: 144B = 16B-aligned, conflict-free b128 loads
#define NW   8    // waves per block (256 threads)

// Two 8-half runs (p[0..7], p[16..23]) -> v16h via two aligned b128 LDS loads.
__device__ __forceinline__ v16h load16(const _Float16* p) {
  const h8* q = (const h8*)p;
  h8 lo = q[0], hi = q[2];
  return __builtin_shufflevector(lo, hi, 0,1,2,3,4,5,6,7,8,9,10,11,12,13,14,15);
}

__global__ __launch_bounds__(256) void node_rk4_kernel(
    const float* __restrict__ x,    const float* __restrict__ W_in,
    const float* __restrict__ b_in, const float* __restrict__ Wf,
    const float* __restrict__ bf,   const float* __restrict__ W_out,
    const float* __restrict__ b_out, float* __restrict__ out)
{
  // Weights staged TRANSPOSED [N][K] in f16 so each B-fragment (fixed N, contiguous K)
  // is two aligned ds_load_b128. winT is K-padded to 32 with zeros (no divergence).
  __shared__ alignas(16) _Float16 wfhT [64 * 64];  // [n][k], k contiguous
  __shared__ alignas(16) _Float16 winT [64 * 32];  // [n][k], k=16..31 zero
  __shared__ alignas(16) _Float16 woutT[16 * 64];  // [n][k]
  __shared__ alignas(16) _Float16 scr  [NW][16 * ROWH];

  const int t = threadIdx.x;
  for (int i = t; i < 64 * 32; i += 256) winT[i] = (_Float16)0.0f;
  __syncthreads();
  for (int i = t; i < 64 * 64; i += 256) {
    int k = i >> 6, n = i & 63;
    wfhT[n * 64 + k] = (_Float16)Wf[i];
  }
  for (int i = t; i < 16 * 64; i += 256) {
    int k = i >> 6, n = i & 63;
    winT[n * 32 + k] = (_Float16)W_in[i];
  }
  for (int i = t; i < 64 * 16; i += 256) {
    int k = i >> 4, n = i & 15;
    woutT[n * 64 + k] = (_Float16)W_out[i];
  }
  __syncthreads();

  const int wave = t >> 5;
  const int lane = t & 31;
  const int hf   = lane >> 4;   // half-wave: 0 or 1
  const int ln   = lane & 15;
  const long tile = (long)blockIdx.x * NW + wave;
  _Float16* myscr = scr[wave];

  // Per-lane bias scalars (C/D layout: column = ln within each 16-wide N tile).
  float binl[4], bfl[4];
  #pragma unroll
  for (int nt = 0; nt < 4; ++nt) {
    binl[nt] = b_in[16 * nt + ln];
    bfl[nt]  = bf  [16 * nt + ln];
  }
  const float boutl = b_out[ln];

  // Wf as resident B-fragments: 2 K-tiles x 4 N-tiles of 32x16 f16.
  // B layout (16-bit, 32x16): lane l holds column N=l%16, element e -> K = 16*(e/8) + 8*(l/16) + e%8.
  v16h wB[2][4];
  #pragma unroll
  for (int kt = 0; kt < 2; ++kt)
    #pragma unroll
    for (int nt = 0; nt < 4; ++nt)
      wB[kt][nt] = load16(wfhT + (16 * nt + ln) * 64 + 32 * kt + 8 * hf);

  // ---- input layer: y0 = x @ W_in + b_in (K=16 zero-padded to one K=32 WMMA) ----
  // A layout (16-bit, 16x32): lane l holds row M=l%16; element e -> K = 16*(e/8) + 8*(l/16) + e%8.
  v16h xa;
  {
    const float* xr = x + (tile * 16 + ln) * 16 + 8 * hf;  // 8 contiguous f32
    #pragma unroll
    for (int e = 0; e < 8; ++e) xa[e] = (_Float16)xr[e];   // K = 8*hf + e  (< 16)
    #pragma unroll
    for (int e = 8; e < 16; ++e) xa[e] = (_Float16)0.0f;   // K >= 16 -> zero pad
  }
  v8f y[4];
  #pragma unroll
  for (int nt = 0; nt < 4; ++nt) {
    v16h wb = load16(winT + (16 * nt + ln) * 32 + 8 * hf); // K>=16 rows are zero in LDS
    y[nt] = __builtin_amdgcn_wmma_f32_16x16x32_f16(false, xa, false, wb,
                                                   (short)0, (v8f){}, false, false);
    #pragma unroll
    for (int i = 0; i < 8; ++i) y[nt][i] += binl[nt];
  }

  // ---- RK4 integration, 32 steps ----
  const float dt = 1.0f / 32.0f, hdt = 0.5f / 32.0f, dt6 = dt / 6.0f;
  #pragma unroll 1
  for (int s = 0; s < 32; ++s) {
    v8f ynew[4], cur[4];
    #pragma unroll
    for (int nt = 0; nt < 4; ++nt) { ynew[nt] = y[nt]; cur[nt] = y[nt]; }

    #pragma unroll
    for (int sub = 0; sub < 4; ++sub) {
      // Packed f32->f16 row-pair converts into independent regs (v_cvt_pk_f16_f32).
      h2 pk[4][4];
      #pragma unroll
      for (int nt = 0; nt < 4; ++nt)
        #pragma unroll
        for (int j = 0; j < 4; ++j) {
          f2 f = { cur[nt][2 * j], cur[nt][2 * j + 1] };
          pk[nt][j] = __builtin_convertvector(f, h2);
        }
      // C-layout -> canonical row-major f16 LDS (row = v + 8*hf, col = 16*nt + ln);
      // even row stores .l (ds_store_b16), odd row stores .h (ds_store_b16_d16_hi).
      #pragma unroll
      for (int nt = 0; nt < 4; ++nt)
        #pragma unroll
        for (int j = 0; j < 4; ++j) {
          _Float16* p = myscr + (2 * j + 8 * hf) * ROWH + 16 * nt + ln;
          p[0]    = pk[nt][j][0];
          p[ROWH] = pk[nt][j][1];
        }

      // A-fragments: per K-tile two contiguous 8-half runs (K=32kt+8hf+0..7 and +16..23) -> 2x b128
      v16h a0, a1;
      {
        const h8* p0 = (const h8*)(myscr + ln * ROWH + 8 * hf);
        a0 = __builtin_shufflevector(p0[0], p0[2], 0,1,2,3,4,5,6,7,8,9,10,11,12,13,14,15);
        const h8* p1 = (const h8*)(myscr + ln * ROWH + 32 + 8 * hf);
        a1 = __builtin_shufflevector(p1[0], p1[2], 0,1,2,3,4,5,6,7,8,9,10,11,12,13,14,15);
      }

      const float wk    = ((sub == 1 || sub == 2) ? 2.0f : 1.0f) * dt6;  // dt/6 * RK4 weight
      const float cstep = (sub < 2) ? hdt : dt;

      #pragma unroll
      for (int nt = 0; nt < 4; ++nt) {
        // z = A*Wf (+0 inline C, no register copies), bias added in epilogue
        v8f c = __builtin_amdgcn_wmma_f32_16x16x32_f16(false, a0, false, wB[0][nt],
                                                       (short)0, (v8f){}, false, false);
        c = __builtin_amdgcn_wmma_f32_16x16x32_f16(false, a1, false, wB[1][nt],
                                                   (short)0, c, false, false);
        #pragma unroll
        for (int i = 0; i < 8; ++i) {
          float k = TANHF(c[i] + bfl[nt]);
          ynew[nt][i] += wk * k;
          if (sub < 3) cur[nt][i] = y[nt][i] + cstep * k;
        }
      }
    }
    #pragma unroll
    for (int nt = 0; nt < 4; ++nt) y[nt] = ynew[nt];   // SSA rename, no copies
  }

  // ---- readout: out = y1 @ W_out + b_out (16x64 @ 64x16, 2 WMMAs) ----
  #pragma unroll
  for (int nt = 0; nt < 4; ++nt)
    #pragma unroll
    for (int j = 0; j < 4; ++j) {
      f2 f = { y[nt][2 * j], y[nt][2 * j + 1] };
      h2 p2 = __builtin_convertvector(f, h2);
      _Float16* p = myscr + (2 * j + 8 * hf) * ROWH + 16 * nt + ln;
      p[0]    = p2[0];
      p[ROWH] = p2[1];
    }

  v16h a0, a1;
  {
    const h8* p0 = (const h8*)(myscr + ln * ROWH + 8 * hf);
    a0 = __builtin_shufflevector(p0[0], p0[2], 0,1,2,3,4,5,6,7,8,9,10,11,12,13,14,15);
    const h8* p1 = (const h8*)(myscr + ln * ROWH + 32 + 8 * hf);
    a1 = __builtin_shufflevector(p1[0], p1[2], 0,1,2,3,4,5,6,7,8,9,10,11,12,13,14,15);
  }
  v16h wo0 = load16(woutT + ln * 64 + 8 * hf);
  v16h wo1 = load16(woutT + ln * 64 + 32 + 8 * hf);

  v8f o = __builtin_amdgcn_wmma_f32_16x16x32_f16(false, a0, false, wo0,
                                                 (short)0, (v8f){}, false, false);
  o = __builtin_amdgcn_wmma_f32_16x16x32_f16(false, a1, false, wo1,
                                             (short)0, o, false, false);

  #pragma unroll
  for (int v = 0; v < 8; ++v)
    out[(tile * 16 + v + 8 * hf) * 16 + ln] = o[v] + boutl;
}

extern "C" void kernel_launch(void* const* d_in, const int* in_sizes, int n_in,
                              void* d_out, int out_size, void* d_ws, size_t ws_size,
                              hipStream_t stream) {
  const float* x     = (const float*)d_in[0];
  const float* W_in  = (const float*)d_in[1];
  const float* b_in  = (const float*)d_in[2];
  const float* Wf    = (const float*)d_in[3];
  const float* bf    = (const float*)d_in[4];
  const float* W_out = (const float*)d_in[5];
  const float* b_out = (const float*)d_in[6];
  float* out = (float*)d_out;

  const int batch  = in_sizes[0] / 16;     // [B, IN_DIM=16]
  const int tiles  = batch / 16;           // 16 rows per wave
  const int blocks = tiles / NW;           // 8 waves per block
  node_rk4_kernel<<<blocks, 256, 0, stream>>>(x, W_in, b_in, Wf, bf, W_out, b_out, out);
}
